// MultiHeadAttention_22325240005233
// MI455X (gfx1250) — matrix operand order repaired and checked by reading, not compiled, not run on hardware
//
#include <hip/hip_runtime.h>

#define BB 2
#define SS 2048
#define EE 1024
#define HH 16
#define AA 1024
#define DHH 64

typedef __attribute__((ext_vector_type(16))) __bf16 v16bf;
typedef __attribute__((ext_vector_type(8)))  __bf16 v8bf;
typedef __attribute__((ext_vector_type(8)))  float  v8f;

union ABFrag {
    v16bf v;
    v8bf  h[2];
};

static __device__ __forceinline__ v8f wmma_bf16(v16bf a, v16bf b, v8f c) {
    return __builtin_amdgcn_wmma_f32_16x16x32_bf16(
        false, a, false, b, (short)0, c, false, false);
}

// A fragment: per-lane two 16B chunks at +0 and +8 elements
static __device__ __forceinline__ void loadA(const __bf16* p, ABFrag& f) {
    f.h[0] = *(const v8bf*)(p);
    f.h[1] = *(const v8bf*)(p + 8);
}
// B fragment: per-lane 32 contiguous bytes
static __device__ __forceinline__ void loadB(const __bf16* p, ABFrag& f) {
    f.h[0] = *(const v8bf*)(p);
    f.h[1] = *(const v8bf*)(p + 8);
}

// CDNA5 async global->LDS copy (16B per lane), tracked by ASYNCcnt
static __device__ __forceinline__ void async16(__bf16* lds, const __bf16* g) {
    const unsigned l = (unsigned)(size_t)lds;           // low 32 bits = LDS offset
    asm volatile("global_load_async_to_lds_b128 %0, %1, off"
                 :: "v"(l), "v"((unsigned long long)(size_t)g)
                 : "memory");
}
static __device__ __forceinline__ void wait_async0() {
    asm volatile("s_wait_asynccnt 0x0" ::: "memory");
}

// ---------------- pre-pass: f32 -> bf16 conversions ----------------

__global__ void k_cvt_x(const float* __restrict__ x, __bf16* __restrict__ y, int n) {
    for (int i = blockIdx.x * blockDim.x + threadIdx.x; i < n; i += gridDim.x * blockDim.x)
        y[i] = (__bf16)x[i];
}

// [h][e][d] (f32) -> [h][d][e] (bf16)
__global__ void k_cvt_wT(const float* __restrict__ w, __bf16* __restrict__ wT,
                         int nh, int ne, int nd) {
    int n = nh * ne * nd;
    for (int i = blockIdx.x * blockDim.x + threadIdx.x; i < n; i += gridDim.x * blockDim.x) {
        int h = i / (ne * nd);
        int r = i - h * (ne * nd);
        int e = r / nd;
        int d = r - e * nd;
        wT[((size_t)h * nd + d) * ne + e] = (__bf16)w[i];
    }
}

// ---------------- phase 1: QKV projection GEMMs ----------------
__global__ void __launch_bounds__(128)
k_qkv_gemm(const __bf16* __restrict__ xbf,
           const __bf16* __restrict__ wqT, const __bf16* __restrict__ wkT,
           const __bf16* __restrict__ wvT,
           const float* __restrict__ bq, const float* __restrict__ bk,
           const float* __restrict__ bv,
           __bf16* __restrict__ qo, __bf16* __restrict__ ko,
           __bf16* __restrict__ vTo) {
    const int mi = blockIdx.z;  // 0=q 1=k 2=v
    const __bf16* wT  = (mi == 0) ? wqT : (mi == 1) ? wkT : wvT;
    const float* bias = (mi == 0) ? bq  : (mi == 1) ? bk  : bv;
    __bf16* out       = (mi == 0) ? qo  : (mi == 1) ? ko  : vTo;

    const int bh = blockIdx.y, b = bh >> 4, h = bh & 15;
    const int wid = threadIdx.x >> 5, lane = threadIdx.x & 31;
    const int lrow = lane & 15, half = lane >> 4;
    const int m0 = (blockIdx.x * 4 + wid) * 32;

    const __bf16* a0 = xbf + (size_t)b * SS * EE + (size_t)(m0 + lrow) * EE + half * 16;
    const __bf16* a1 = a0 + (size_t)16 * EE;
    const __bf16* wrow = wT + (size_t)h * DHH * EE + (size_t)lrow * EE + half * 16;

    v8f acc[2][4] = {};
    ABFrag aC[2], bC[4], aN[2], bN[4];

#define QKV_LOAD(KC, AF, BF)                                        \
    do {                                                            \
        loadA(a0 + (KC) * 32, (AF)[0]);                             \
        loadA(a1 + (KC) * 32, (AF)[1]);                             \
        _Pragma("unroll") for (int nt = 0; nt < 4; ++nt)            \
            loadB(wrow + (size_t)nt * 16 * EE + (KC) * 32, (BF)[nt]); \
    } while (0)
#define QKV_MATH(AF, BF)                                            \
    do {                                                            \
        _Pragma("unroll") for (int mt = 0; mt < 2; ++mt)            \
            _Pragma("unroll") for (int nt = 0; nt < 4; ++nt)        \
                acc[mt][nt] = wmma_bf16((AF)[mt].v, (BF)[nt].v, acc[mt][nt]); \
    } while (0)

    const int NC = EE / 32;
    QKV_LOAD(0, aC, bC);
    for (int kc = 0; kc + 2 < NC; kc += 2) {
        QKV_LOAD(kc + 1, aN, bN);
        QKV_MATH(aC, bC);
        QKV_LOAD(kc + 2, aC, bC);
        QKV_MATH(aN, bN);
    }
    QKV_LOAD(NC - 1, aN, bN);
    QKV_MATH(aC, bC);
    QKV_MATH(aN, bN);
#undef QKV_LOAD
#undef QKV_MATH

    const int srow = half * 8;
#pragma unroll
    for (int mt = 0; mt < 2; ++mt) {
#pragma unroll
        for (int nt = 0; nt < 4; ++nt) {
            const float vb = bias[h * DHH + nt * 16 + lrow];
#pragma unroll
            for (int r = 0; r < 8; ++r) {
                const float val = acc[mt][nt][r] + vb;
                const int row = m0 + mt * 16 + r + srow;
                if (mi < 2)
                    out[((size_t)bh * SS + row) * DHH + nt * 16 + lrow] = (__bf16)val;
                else  // v stored transposed: [bh][dh][s]
                    out[((size_t)bh * DHH + nt * 16 + lrow) * SS + row] = (__bf16)val;
            }
        }
    }
}

// ---------------- phase 2: flash attention ----------------
// Block = 4 waves x 16 queries. k/v blocks staged once per workgroup into LDS
// via async global->LDS (double-buffered), shared by all 4 waves.
__global__ void __launch_bounds__(128)
k_attn(const __bf16* __restrict__ q, const __bf16* __restrict__ kmat,
       const __bf16* __restrict__ vT, __bf16* __restrict__ concat) {
    // K stage: 32 keys x 64 dh, padded row stride 72 elem (36-bank step)
    __shared__ __align__(16) __bf16 kstage[2][32 * 72];
    // V stage: 64 dh rows x 32 keys, padded row stride 40 elem (20-bank step)
    __shared__ __align__(16) __bf16 vstage[2][64 * 40];
    // per-wave p-tile bounce (C-layout -> A-layout), 80B row stride
    __shared__ __align__(16) __bf16 pbuf[4][16 * 40];

    const int bh = blockIdx.y, b = bh >> 4, h = bh & 15;
    const int tid = threadIdx.x;
    const int wid = tid >> 5, lane = tid & 31;
    const int lrow = lane & 15, half = lane >> 4;
    const int m0 = (blockIdx.x * 4 + wid) * 16;

    const __bf16* qp = q    + (size_t)bh * SS * DHH;
    const __bf16* kp = kmat + (size_t)bh * SS * DHH;
    const __bf16* vp = vT   + (size_t)bh * DHH * SS;

    // cooperative async stage of one 32-key block of k and v
#define STAGE_KV(T0, P)                                                          \
    do {                                                                         \
        _Pragma("unroll") for (int i = 0; i < 2; ++i) {                          \
            const int c = tid * 2 + i;                                           \
            const int row = c >> 3, off = (c & 7) * 8;                           \
            async16(&kstage[(P)][row * 72 + off],                                \
                    kp + (size_t)((T0) + row) * DHH + off);                      \
        }                                                                        \
        _Pragma("unroll") for (int i = 0; i < 2; ++i) {                          \
            const int c = tid * 2 + i;                                           \
            const int row = c >> 2, off = (c & 3) * 8;                           \
            async16(&vstage[(P)][row * 40 + off],                                \
                    vp + (size_t)row * SS + (T0) + off);                         \
        }                                                                        \
    } while (0)

    // q A-fragments (M=16, K=64 in two 32-chunks), resident all pass
    ABFrag qa[2];
#pragma unroll
    for (int c = 0; c < 2; ++c)
        loadA(qp + (size_t)(m0 + lrow) * DHH + c * 32 + half * 16, qa[c]);

    v8f acc[4] = {};
    float m8[8], l8[8];
#pragma unroll
    for (int r = 0; r < 8; ++r) { m8[r] = -1e30f; l8[r] = 0.0f; }

    STAGE_KV(0, 0);
    wait_async0();
    __syncthreads();

    for (int t0 = 0; t0 < SS; t0 += 32) {
        const int p = (t0 >> 5) & 1;
        if (t0 + 32 < SS) STAGE_KV(t0 + 32, p ^ 1);  // overlap staging with compute

        // k B-fragments from LDS: [0]=(c0,keys 0-15) [1]=(c0,16-31) [2]=(c1,0-15) [3]=(c1,16-31)
        ABFrag kf[4];
        loadB(&kstage[p][lrow * 72 + half * 16], kf[0]);
        loadB(&kstage[p][(lrow + 16) * 72 + half * 16], kf[1]);
        loadB(&kstage[p][lrow * 72 + 32 + half * 16], kf[2]);
        loadB(&kstage[p][(lrow + 16) * 72 + 32 + half * 16], kf[3]);

        v8f s0 = {}, s1 = {};
        s0 = wmma_bf16(qa[0].v, kf[0].v, s0);
        s1 = wmma_bf16(qa[0].v, kf[1].v, s1);
        s0 = wmma_bf16(qa[1].v, kf[2].v, s0);
        s1 = wmma_bf16(qa[1].v, kf[3].v, s1);

        // v B-fragments from LDS (issued before softmax VALU to cover latency)
        ABFrag vf[4];
#pragma unroll
        for (int nt = 0; nt < 4; ++nt)
            loadB(&vstage[p][(nt * 16 + lrow) * 40 + half * 16], vf[nt]);

        // online softmax; row r lives in element r, row index r + 8*half
        float mn[8];
#pragma unroll
        for (int r = 0; r < 8; ++r) {
            const float sa = s0[r] * 0.125f, sb = s1[r] * 0.125f;
            s0[r] = sa; s1[r] = sb;
            float t = fmaxf(sa, sb);
            t = fmaxf(t, __shfl_xor(t, 1, 32));
            t = fmaxf(t, __shfl_xor(t, 2, 32));
            t = fmaxf(t, __shfl_xor(t, 4, 32));
            t = fmaxf(t, __shfl_xor(t, 8, 32));
            mn[r] = fmaxf(m8[r], t);
        }
#pragma unroll
        for (int r = 0; r < 8; ++r) {
            const float al = __expf(m8[r] - mn[r]);
            const float p0 = __expf(s0[r] - mn[r]);
            const float p1 = __expf(s1[r] - mn[r]);
            s0[r] = p0; s1[r] = p1;
            float ls = p0 + p1;
            ls += __shfl_xor(ls, 1, 32);
            ls += __shfl_xor(ls, 2, 32);
            ls += __shfl_xor(ls, 4, 32);
            ls += __shfl_xor(ls, 8, 32);
            l8[r] = l8[r] * al + ls;
            m8[r] = mn[r];
            acc[0][r] *= al; acc[1][r] *= al;
            acc[2][r] *= al; acc[3][r] *= al;
        }
        // C-layout -> A-layout bounce through per-wave LDS tile
        const int sro = half * 8;
#pragma unroll
        for (int r = 0; r < 8; ++r) {
            pbuf[wid][(r + sro) * 40 + lrow]      = (__bf16)s0[r];
            pbuf[wid][(r + sro) * 40 + 16 + lrow] = (__bf16)s1[r];
        }
        asm volatile("s_wait_dscnt 0x0" ::: "memory");
        ABFrag pa;
        loadA(&pbuf[wid][lrow * 40 + half * 16], pa);

#pragma unroll
        for (int nt = 0; nt < 4; ++nt)
            acc[nt] = wmma_bf16(pa.v, vf[nt].v, acc[nt]);

        wait_async0();      // own async stage writes done
        __syncthreads();    // everyone's stage writes visible; stage buffer reusable
    }
#undef STAGE_KV

    // epilogue: normalize, store to concat [b][s][h*DH + d]
    const int srow = half * 8;
#pragma unroll
    for (int r = 0; r < 8; ++r) {
        const float inv = 1.0f / l8[r];
        const size_t row = (size_t)b * SS + m0 + r + srow;
#pragma unroll
        for (int nt = 0; nt < 4; ++nt)
            concat[row * AA + h * DHH + nt * 16 + lrow] = (__bf16)(acc[nt][r] * inv);
    }
}

// ---------------- phase 3: output projection ----------------
__global__ void __launch_bounds__(128)
k_out_gemm(const __bf16* __restrict__ concat, const __bf16* __restrict__ woT,
           const float* __restrict__ bo, float* __restrict__ out) {
    const int wid = threadIdx.x >> 5, lane = threadIdx.x & 31;
    const int lrow = lane & 15, half = lane >> 4;
    const int m0 = (blockIdx.y * 4 + wid) * 32;
    const int n0 = blockIdx.x * 64;

    const __bf16* a0 = concat + (size_t)(m0 + lrow) * AA + half * 16;
    const __bf16* a1 = a0 + (size_t)16 * AA;
    const __bf16* wrow = woT + (size_t)(n0 + lrow) * AA + half * 16;

    v8f acc[2][4] = {};
    ABFrag aC[2], bC[4], aN[2], bN[4];

#define OUT_LOAD(KC, AF, BF)                                        \
    do {                                                            \
        loadA(a0 + (KC) * 32, (AF)[0]);                             \
        loadA(a1 + (KC) * 32, (AF)[1]);                             \
        _Pragma("unroll") for (int nt = 0; nt < 4; ++nt)            \
            loadB(wrow + (size_t)nt * 16 * AA + (KC) * 32, (BF)[nt]); \
    } while (0)
#define OUT_MATH(AF, BF)                                            \
    do {                                                            \
        _Pragma("unroll") for (int mt = 0; mt < 2; ++mt)            \
            _Pragma("unroll") for (int nt = 0; nt < 4; ++nt)        \
                acc[mt][nt] = wmma_bf16((AF)[mt].v, (BF)[nt].v, acc[mt][nt]); \
    } while (0)

    const int NC = AA / 32;
    OUT_LOAD(0, aC, bC);
    for (int kc = 0; kc + 2 < NC; kc += 2) {
        OUT_LOAD(kc + 1, aN, bN);
        OUT_MATH(aC, bC);
        OUT_LOAD(kc + 2, aC, bC);
        OUT_MATH(aN, bN);
    }
    OUT_LOAD(NC - 1, aN, bN);
    OUT_MATH(aC, bC);
    OUT_MATH(aN, bN);
#undef OUT_LOAD
#undef OUT_MATH

    const int srow = half * 8;
#pragma unroll
    for (int mt = 0; mt < 2; ++mt) {
#pragma unroll
        for (int nt = 0; nt < 4; ++nt) {
            const float vb = bo[n0 + nt * 16 + lrow];
#pragma unroll
            for (int r = 0; r < 8; ++r)
                out[(size_t)(m0 + mt * 16 + r + srow) * EE + n0 + nt * 16 + lrow] =
                    acc[mt][nt][r] + vb;
        }
    }
}

// ---------------- launch ----------------
extern "C" void kernel_launch(void* const* d_in, const int* in_sizes, int n_in,
                              void* d_out, int out_size, void* d_ws, size_t ws_size,
                              hipStream_t stream) {
    (void)in_sizes; (void)n_in; (void)out_size; (void)ws_size;
    const float* x  = (const float*)d_in[0];
    const float* Wq = (const float*)d_in[1];
    const float* bq = (const float*)d_in[2];
    const float* Wk = (const float*)d_in[3];
    const float* bk = (const float*)d_in[4];
    const float* Wv = (const float*)d_in[5];
    const float* bv = (const float*)d_in[6];
    const float* Wo = (const float*)d_in[7];
    const float* bo = (const float*)d_in[8];
    float* out = (float*)d_out;

    __bf16* w = (__bf16*)d_ws;
    size_t off = 0;
    __bf16* xbf    = w + off; off += (size_t)BB * SS * EE;
    __bf16* wqT    = w + off; off += (size_t)HH * DHH * EE;
    __bf16* wkT    = w + off; off += (size_t)HH * DHH * EE;
    __bf16* wvT    = w + off; off += (size_t)HH * DHH * EE;
    __bf16* woT    = w + off; off += (size_t)EE * AA;
    __bf16* qb     = w + off; off += (size_t)BB * HH * SS * DHH;
    __bf16* kb     = w + off; off += (size_t)BB * HH * SS * DHH;
    __bf16* vTb    = w + off; off += (size_t)BB * HH * SS * DHH;
    __bf16* concat = w + off; off += (size_t)BB * SS * AA;

    {
        const int n = BB * SS * EE;
        k_cvt_x<<<(n + 255) / 256, 256, 0, stream>>>(x, xbf, n);
    }
    k_cvt_wT<<<4096, 256, 0, stream>>>(Wq, wqT, HH, EE, DHH);
    k_cvt_wT<<<4096, 256, 0, stream>>>(Wk, wkT, HH, EE, DHH);
    k_cvt_wT<<<4096, 256, 0, stream>>>(Wv, wvT, HH, EE, DHH);
    k_cvt_wT<<<4096, 256, 0, stream>>>(Wo, woT, 1, AA, EE);  // [A,E] -> [E,A]

    k_qkv_gemm<<<dim3(SS / 128, BB * HH, 3), 128, 0, stream>>>(
        xbf, wqT, wkT, wvT, bq, bk, bv, qb, kb, vTb);

    k_attn<<<dim3(SS / 64, BB * HH), 128, 0, stream>>>(qb, kb, vTb, concat);

    k_out_gemm<<<dim3(EE / 64, (BB * SS) / 128), 128, 0, stream>>>(concat, woT, bo, out);
}
